// WindowAttention_57166014710352
// MI455X (gfx1250) — matrix operand order repaired and checked by reading, not compiled, not run on hardware
//
#include <hip/hip_runtime.h>

// MI455X / gfx1250: wave32, WMMA 16x16x32 f16 (fp32 accum).
// Pipeline: cvt->f16, qkv GEMMs (WMMA), fused attention (WMMA qk^T + softmax + WMMA pV),
// proj GEMM (WMMA, +bias, fp32 out).
// Workspace requirement: ~303 MB (see offsets in kernel_launch).

typedef _Float16 half_t;
typedef __attribute__((ext_vector_type(16))) _Float16 v16h;
typedef __attribute__((ext_vector_type(8)))  float    v8f;

#define DEV static __device__ __forceinline__

DEV v8f wmma16(v16h a, v16h b, v8f c) {
  // D = A(16x32 f16) * B(32x16 f16) + C(16x16 f32)
  return __builtin_amdgcn_wmma_f32_16x16x32_f16(false, a, false, b, (short)0, c, false, false);
}

// A-fragment: lane holds row (m = lane&15); element e -> K = (e>>3)*16 + hi*8 + (e&7)
// => two contiguous 16B chunks at k0+hi*8 and k0+16+hi*8 of a K-major row.
DEV v16h load_a(const half_t* base, int ld, int row0, int k0) {
  const int lane = threadIdx.x & 31;
  const int r = lane & 15, hi = lane >> 4;
  const half_t* p = base + (size_t)(row0 + r) * ld + k0 + hi * 8;
  union { uint4 u[2]; v16h v; } t;
  t.u[0] = *reinterpret_cast<const uint4*>(p);
  t.u[1] = *reinterpret_cast<const uint4*>(p + 16);
  return t.v;
}

// B-fragment: lane holds column (n = lane&15); element e -> K = (e>>3)*16 + hi*8 + (e&7)
// => two contiguous 16B chunks at k0+hi*8 and k0+16+hi*8 of a K-major row (Bmat[k][n] = W[n][k]).
DEV v16h load_b(const half_t* base, int ld, int col0, int k0) {
  const int lane = threadIdx.x & 31;
  const int c = lane & 15, hi = lane >> 4;
  const half_t* p = base + (size_t)(col0 + c) * ld + k0 + hi * 8;
  union { uint4 u[2]; v16h v; } t;
  t.u[0] = *reinterpret_cast<const uint4*>(p);
  t.u[1] = *reinterpret_cast<const uint4*>(p + 16);
  return t.v;
}

// ---------------- fp32 -> fp16 conversion ----------------
__global__ void cvt_f32_f16(const float* __restrict__ s, half_t* __restrict__ d, int n) {
  int i = blockIdx.x * blockDim.x + threadIdx.x;
  int stride = gridDim.x * blockDim.x;
  for (; i < n; i += stride) d[i] = (half_t)s[i];
}

// ---------------- GEMM: C[M][N](f16) = A[M][K] * B[N][K]^T ----------------
// block = 128 threads (4 waves); each wave -> 64x64 tile; WG -> 128x128.
__global__ void __launch_bounds__(128) gemm_f16out(const half_t* __restrict__ A,
                                                   const half_t* __restrict__ B,
                                                   half_t* __restrict__ C,
                                                   int K, int ldc) {
  const int wave = threadIdx.x >> 5;
  const int lane = threadIdx.x & 31;
  const int hi = lane >> 4, cl = lane & 15;
  const int m0 = blockIdx.x * 128 + (wave >> 1) * 64;
  const int n0 = blockIdx.y * 128 + (wave & 1) * 64;

  v8f zf = {};
  v8f acc[4][4];
#pragma unroll
  for (int i = 0; i < 4; ++i)
#pragma unroll
    for (int j = 0; j < 4; ++j) acc[i][j] = zf;

  for (int k0 = 0; k0 < K; k0 += 32) {
    if (k0 + 32 < K)  // hint next A tile into cache (global_prefetch_b8)
      __builtin_prefetch(A + (size_t)m0 * K + k0 + 32, 0, 0);
    v16h a[4];
#pragma unroll
    for (int i = 0; i < 4; ++i) a[i] = load_a(A, K, m0 + i * 16, k0);
#pragma unroll
    for (int j = 0; j < 4; ++j) {
      v16h b = load_b(B, K, n0 + j * 16, k0);
#pragma unroll
      for (int i = 0; i < 4; ++i) acc[i][j] = wmma16(a[i], b, acc[i][j]);
    }
  }
#pragma unroll
  for (int i = 0; i < 4; ++i)
#pragma unroll
    for (int j = 0; j < 4; ++j) {
      const size_t row = (size_t)(m0 + i * 16 + 8 * hi);
      const int col = n0 + j * 16 + cl;
#pragma unroll
      for (int r = 0; r < 8; ++r)
        C[(row + r) * (size_t)ldc + col] = (half_t)acc[i][j][r];
    }
}

// ---------------- GEMM with bias, fp32 out (final projection) ----------------
__global__ void __launch_bounds__(128) gemm_f32out_bias(const half_t* __restrict__ A,
                                                        const half_t* __restrict__ B,
                                                        float* __restrict__ C,
                                                        const float* __restrict__ bias,
                                                        int K, int ldc) {
  const int wave = threadIdx.x >> 5;
  const int lane = threadIdx.x & 31;
  const int hi = lane >> 4, cl = lane & 15;
  const int m0 = blockIdx.x * 128 + (wave >> 1) * 64;
  const int n0 = blockIdx.y * 128 + (wave & 1) * 64;

  v8f zf = {};
  v8f acc[4][4];
#pragma unroll
  for (int i = 0; i < 4; ++i)
#pragma unroll
    for (int j = 0; j < 4; ++j) acc[i][j] = zf;

  for (int k0 = 0; k0 < K; k0 += 32) {
    if (k0 + 32 < K)
      __builtin_prefetch(A + (size_t)m0 * K + k0 + 32, 0, 0);
    v16h a[4];
#pragma unroll
    for (int i = 0; i < 4; ++i) a[i] = load_a(A, K, m0 + i * 16, k0);
#pragma unroll
    for (int j = 0; j < 4; ++j) {
      v16h b = load_b(B, K, n0 + j * 16, k0);
#pragma unroll
      for (int i = 0; i < 4; ++i) acc[i][j] = wmma16(a[i], b, acc[i][j]);
    }
  }
#pragma unroll
  for (int i = 0; i < 4; ++i)
#pragma unroll
    for (int j = 0; j < 4; ++j) {
      const size_t row = (size_t)(m0 + i * 16 + 8 * hi);
      const int col = n0 + j * 16 + cl;
      const float bv = bias[col];
#pragma unroll
      for (int r = 0; r < 8; ++r)
        C[(row + r) * (size_t)ldc + col] = acc[i][j][r] + bv;
    }
}

// ---------------- fused attention ----------------
// One WG per (b, h). N = seq length (128 self / 64 mutual). N/16 waves; wave w owns
// query rows [w*16, w*16+16). qkv layout: row-major [B*128][768], cols q|k|v each h*32.
// Output written into concat buffer xc [B*128][512] at (out_row_base, out_col_base+h*32).
template <int N>
__global__ void __launch_bounds__(N * 2) attn_kernel(const half_t* __restrict__ qkv,
                                                     const float* __restrict__ mask,
                                                     half_t* __restrict__ xc,
                                                     int q_base, int kv_base,
                                                     int out_row_base, int out_col_base) {
  constexpr int JT = N / 16;
  constexpr int LD = N + 8;
  constexpr float SCALE = 0.17677669529663687f;  // 32^-0.5

  __shared__ half_t Vt[32][LD];   // V transposed: Vt[d][m]
  __shared__ half_t P[N][LD];     // softmax probabilities (f16)

  const int tid = threadIdx.x;
  const int wave = tid >> 5;
  const int lane = tid & 31;
  const int hi = lane >> 4, cl = lane & 15;
  const int b = blockIdx.x >> 3;
  const int h = blockIdx.x & 7;
  const size_t qkv_row0 = (size_t)b * 128;

  // Stage V transposed into LDS.
  const half_t* vptr = qkv + (qkv_row0 + kv_base) * 768 + 512 + h * 32;
  for (int idx = tid; idx < N * 32; idx += blockDim.x) {
    const int i = idx >> 5, dd = idx & 31;
    Vt[dd][i] = vptr[(size_t)i * 768 + dd];
  }
  __syncthreads();

  // S = q . k^T  (K-dim = d = 32 -> single WMMA per 16x16 tile)
  const half_t* qptr = qkv + (qkv_row0 + q_base) * 768 + h * 32;
  const half_t* kptr = qkv + (qkv_row0 + kv_base) * 768 + 256 + h * 32;
  const v16h aq = load_a(qptr, 768, wave * 16, 0);
  v8f zf = {};
  v8f sc[JT];
#pragma unroll
  for (int jt = 0; jt < JT; ++jt) {
    v16h bk = load_b(kptr, 768, jt * 16, 0);
    sc[jt] = wmma16(aq, bk, zf);
  }

  // scale + mask + row softmax (rows live in one 16-lane half -> shfl_xor 1/2/4/8)
  const float* mwin = mask + (size_t)(b & 63) * 128 * 128;
#pragma unroll
  for (int r = 0; r < 8; ++r) {
    const int irow = wave * 16 + r + 8 * hi;        // relative query index
    const float* mr = mwin + (size_t)irow * 128 + cl;
    float vals[JT];
    float mx = -1e30f;
#pragma unroll
    for (int jt = 0; jt < JT; ++jt) {
      float v = sc[jt][r] * SCALE + mr[jt * 16];
      vals[jt] = v;
      mx = fmaxf(mx, v);
    }
    mx = fmaxf(mx, __shfl_xor(mx, 1));
    mx = fmaxf(mx, __shfl_xor(mx, 2));
    mx = fmaxf(mx, __shfl_xor(mx, 4));
    mx = fmaxf(mx, __shfl_xor(mx, 8));
    float sum = 0.f;
#pragma unroll
    for (int jt = 0; jt < JT; ++jt) {
      float e = __expf(vals[jt] - mx);
      vals[jt] = e;
      sum += e;
    }
    sum += __shfl_xor(sum, 1);
    sum += __shfl_xor(sum, 2);
    sum += __shfl_xor(sum, 4);
    sum += __shfl_xor(sum, 8);
    const float inv = 1.f / sum;
#pragma unroll
    for (int jt = 0; jt < JT; ++jt)
      P[irow][jt * 16 + cl] = (half_t)(vals[jt] * inv);
  }
  __syncthreads();

  // O = P . V  (A = P strip from LDS, B = Vt from LDS), 2 d-tiles of 16.
#pragma unroll
  for (int ot = 0; ot < 2; ++ot) {
    v8f acc = zf;
#pragma unroll
    for (int mt = 0; mt < N / 32; ++mt) {
      v16h ap = load_a(&P[wave * 16][0], LD, 0, mt * 32);
      v16h bv = load_b(&Vt[0][0], LD, ot * 16, mt * 32);
      acc = wmma16(ap, bv, acc);
    }
    const size_t orow0 = (size_t)b * 128 + out_row_base + wave * 16 + 8 * hi;
    const int ocol = out_col_base + h * 32 + ot * 16 + cl;
#pragma unroll
    for (int r = 0; r < 8; ++r)
      xc[(orow0 + r) * 512 + ocol] = (half_t)acc[r];
  }
}

// ---------------- host launch ----------------
extern "C" void kernel_launch(void* const* d_in, const int* in_sizes, int n_in,
                              void* d_out, int out_size, void* d_ws, size_t ws_size,
                              hipStream_t stream) {
  const float* x    = (const float*)d_in[0];   // [512*128*256]
  const float* mask = (const float*)d_in[1];   // [64*128*128]
  const float* wqs  = (const float*)d_in[2];   // [768*256]
  const float* wqm  = (const float*)d_in[3];   // [768*256]
  const float* wp   = (const float*)d_in[4];   // [256*512]
  const float* bp   = (const float*)d_in[5];   // [256]
  float* out = (float*)d_out;                  // [512*128*256]

  char* ws = (char*)d_ws;
  half_t* x_h   = (half_t*)(ws);               //  33,554,432 B : x as f16
  half_t* wqs_h = (half_t*)(ws +  33554432);   //     393,216 B
  half_t* wqm_h = (half_t*)(ws +  33947648);   //     393,216 B
  half_t* wp_h  = (half_t*)(ws +  34340864);   //     262,144 B
  half_t* qkvS  = (half_t*)(ws +  34603008);   // 100,663,296 B : [65536][768]
  half_t* qkvM  = (half_t*)(ws + 135266304);   // 100,663,296 B
  half_t* xc    = (half_t*)(ws + 235929600);   //  67,108,864 B : [65536][512] = [x_mut|x_out]
  // total: 303,038,464 B

  cvt_f32_f16<<<8192, 256, 0, stream>>>(x,   x_h,   512 * 128 * 256);
  cvt_f32_f16<<<768,  256, 0, stream>>>(wqs, wqs_h, 768 * 256);
  cvt_f32_f16<<<768,  256, 0, stream>>>(wqm, wqm_h, 768 * 256);
  cvt_f32_f16<<<512,  256, 0, stream>>>(wp,  wp_h,  256 * 512);

  // QKV projections: [65536 x 256] @ [256 x 768]
  gemm_f16out<<<dim3(512, 6), 128, 0, stream>>>(x_h, wqs_h, qkvS, 256, 768);
  gemm_f16out<<<dim3(512, 6), 128, 0, stream>>>(x_h, wqm_h, qkvM, 256, 768);

  // Self attention -> xc cols [256,512)
  attn_kernel<128><<<4096, 256, 0, stream>>>(qkvS, mask, xc, 0, 0, 0, 256);
  // Mutual: x1 = attn(q2,k1,v1) -> rows [0,64); x2 = attn(q1,k2,v2) -> rows [64,128); cols [0,256)
  attn_kernel<64><<<4096, 128, 0, stream>>>(qkvM, mask, xc, 64, 0, 0, 0);
  attn_kernel<64><<<4096, 128, 0, stream>>>(qkvM, mask, xc, 0, 64, 64, 0);

  // Projection: [65536 x 512] @ [512 x 256] + bias -> fp32 out
  gemm_f32out_bias<<<dim3(512, 2), 128, 0, stream>>>(xc, wp_h, out, bp, 512, 256);
}
